// Net_23192823398816
// MI455X (gfx1250) — matrix operand order repaired and checked by reading, not compiled, not run on hardware
//
#include <hip/hip_runtime.h>
#include <math.h>

typedef _Float16 h16;
typedef __attribute__((ext_vector_type(2)))  _Float16 h16x2;
typedef __attribute__((ext_vector_type(16))) _Float16 v16h;
typedef __attribute__((ext_vector_type(8)))  float    v8f;

#define DEVI __device__ __forceinline__

constexpr int B_ = 4, N_ = 4096, KNN_ = 16, NBR_ = 64;
constexpr int S1_ = 820, S1A_ = 410, S2_ = 103;

DEVI float leakyf(float x, float a) { return x > 0.f ? x : a * x; }

// ---------------- WMMA fragment helpers (CDNA5 16x16x32 f16 layouts) ----------------
// A (16x32, 16-bit): lane L: row = L&15; k = k0 + (v>>2)*16 + (L>>4)*8 + (v&3)*2
DEVI v16h load_afrag(const h16* buf, int stride, int rowBase, int k0, int lane) {
  int hf = lane >> 4, r = lane & 15;
  const h16* rp = buf + (rowBase + r) * stride + k0 + hf * 8;
  v16h a;
#pragma unroll
  for (int v = 0; v < 8; ++v) {
    int k = ((v >> 2) * 16) + ((v & 3) * 2);
    h16x2 p = *(const h16x2*)(rp + k);
    a[2 * v] = p.x; a[2 * v + 1] = p.y;
  }
  return a;
}
// B (32x16, 16-bit), WT stored transposed (Npad x Kpad): lane L: col = L&15; k = k0 + (v>>2)*16 + (L>>4)*8 + (v&3)*2
DEVI v16h load_bfrag(const h16* wt, int kp, int n0, int k0, int lane) {
  int hf = lane >> 4, c = lane & 15;
  const h16* cp = wt + (size_t)(n0 + c) * kp + k0 + hf * 8;
  v16h b;
#pragma unroll
  for (int v = 0; v < 8; ++v) {
    h16x2 p = *(const h16x2*)(cp + ((v >> 2) * 16) + ((v & 3) * 2));
    b[2 * v] = p.x; b[2 * v + 1] = p.y;
  }
  return b;
}

// One MLP layer (relu(x@W+b)*g+beta) over this wave's 16 rows; in-place LDS update.
template<int KP, int COUT, int STRIDE>
DEVI void mlp_layer(h16* buf, int rowBase, const h16* wt, const float* bias,
                    const float* g, const float* beta, int lane) {
  constexpr int KT = KP / 32, NT = COUT / 16;
  v16h a[KT];
#pragma unroll
  for (int kt = 0; kt < KT; ++kt) a[kt] = load_afrag(buf, STRIDE, rowBase, kt * 32, lane);
  int hf = lane >> 4, c = lane & 15;
  for (int nt = 0; nt < NT; ++nt) {
    v8f d = {};
#pragma unroll
    for (int kt = 0; kt < KT; ++kt) {
      v16h bf = load_bfrag(wt, KP, nt * 16, kt * 32, lane);
      d = __builtin_amdgcn_wmma_f32_16x16x32_f16(false, a[kt], false, bf, (short)0, d, false, false);
    }
    int n = nt * 16 + c;
    float bb = bias[n], gg = g[n], be = beta[n];
#pragma unroll
    for (int r = 0; r < 8; ++r) {
      float h = fmaxf(d[r] + bb, 0.f) * gg + be;
      buf[(size_t)(rowBase + r + 8 * hf) * STRIDE + n] = (h16)h;
    }
  }
}

// Final layer + masked row-max: per-wave partial max written to red[wave*COUT + n]
template<int KP, int COUT, int STRIDE>
DEVI void mlp_final_max(const h16* buf, int rowBase, const h16* wt, const float* bias,
                        const float* g, const float* beta, const int* mask,
                        float* red, int lane, int wave) {
  constexpr int KT = KP / 32, NT = COUT / 16;
  v16h a[KT];
#pragma unroll
  for (int kt = 0; kt < KT; ++kt) a[kt] = load_afrag(buf, STRIDE, rowBase, kt * 32, lane);
  int hf = lane >> 4, c = lane & 15;
  for (int nt = 0; nt < NT; ++nt) {
    v8f d = {};
#pragma unroll
    for (int kt = 0; kt < KT; ++kt) {
      v16h bf = load_bfrag(wt, KP, nt * 16, kt * 32, lane);
      d = __builtin_amdgcn_wmma_f32_16x16x32_f16(false, a[kt], false, bf, (short)0, d, false, false);
    }
    int n = nt * 16 + c;
    float bb = bias[n], gg = g[n], be = beta[n];
    float mx = -INFINITY;
#pragma unroll
    for (int r = 0; r < 8; ++r) {
      float h = fmaxf(d[r] + bb, 0.f) * gg + be;
      h = mask[rowBase + r + 8 * hf] ? h : -INFINITY;
      mx = fmaxf(mx, h);
    }
    mx = fmaxf(mx, __shfl_xor(mx, 16, 32));
    if (hf == 0) red[wave * COUT + n] = mx;
  }
}

// ---------------- Set-abstraction kernel: gather + 3-layer WMMA MLP + masked max ----------------
template<int CX, int CIN, int K0, int C1, int C2, int C3, int STRIDE>
__global__ __launch_bounds__(128) void sa_kernel(
    const float* __restrict__ x, const float* __restrict__ pos, int nsrc,
    const float* __restrict__ q, const int* __restrict__ nidx, int S,
    const h16* wt0, const float* b0, const float* g0, const float* e0,
    const h16* wt1, const float* b1, const float* g1, const float* e1,
    const h16* wt2, const float* b2, const float* g2, const float* e2,
    float* __restrict__ out) {
  __shared__ h16 buf[64 * STRIDE];
  __shared__ int mask[64];
  __shared__ float red[4 * C3];
  int tid = threadIdx.x, lane = tid & 31, wave = tid >> 5;
  int cen = blockIdx.x, b = cen / S;
  const float* xb = x + (size_t)b * nsrc * CX;
  const float* pb = pos + (size_t)b * nsrc * 3;
  const float* qc = q + (size_t)cen * 3;
  const int* nb = nidx + (size_t)cen * NBR_;
  int rowBase = wave * 16;
  int hf = lane >> 4, m = rowBase + (lane & 15);
  int j = nb[m];
  int valid = j >= 0; if (!valid) j = 0;
  if (hf == 0) mask[m] = valid;
  float q0 = qc[0], q1 = qc[1], q2 = qc[2];
  for (int cc = hf; cc < K0; cc += 2) {
    float v = 0.f;
    if (cc < CX)       v = xb[(size_t)j * CX + cc];
    else if (cc < CIN) { int d = cc - CX; v = pb[j * 3 + d] - (d == 0 ? q0 : (d == 1 ? q1 : q2)); }
    buf[(size_t)m * STRIDE + cc] = (h16)v;
  }
  __syncthreads();
  mlp_layer<K0, C1, STRIDE>(buf, rowBase, wt0, b0, g0, e0, lane);
  mlp_layer<C1, C2, STRIDE>(buf, rowBase, wt1, b1, g1, e1, lane);
  mlp_final_max<C2, C3, STRIDE>(buf, rowBase, wt2, b2, g2, e2, mask, red, lane, wave);
  __syncthreads();
  for (int ci = tid; ci < C3; ci += 128) {
    float v = fmaxf(fmaxf(red[ci], red[C3 + ci]), fmaxf(red[2 * C3 + ci], red[3 * C3 + ci]));
    out[(size_t)cen * C3 + ci] = v;
  }
}

// ---------------- Global SA: 32 points per block, partial max per block ----------------
template<int CX, int CIN, int K0, int C1, int C2, int C3, int STRIDE>
__global__ __launch_bounds__(64) void gsa_kernel(
    const float* __restrict__ x, const float* __restrict__ pos, int nsrc, int npts,
    const h16* wt0, const float* b0, const float* g0, const float* e0,
    const h16* wt1, const float* b1, const float* g1, const float* e1,
    const h16* wt2, const float* b2, const float* g2, const float* e2,
    float* __restrict__ partial, int nblk) {
  __shared__ h16 buf[32 * STRIDE];
  __shared__ int mask[32];
  __shared__ float red[2 * C3];
  int tid = threadIdx.x, lane = tid & 31, wave = tid >> 5;
  int b = blockIdx.x / nblk, blk = blockIdx.x % nblk;
  const float* xb = x + (size_t)b * nsrc * CX;
  const float* pb = pos + (size_t)b * nsrc * 3;
  int rowBase = wave * 16;
  int hf = lane >> 4, m = rowBase + (lane & 15);
  int pt = blk * 32 + m;
  int valid = pt < npts; int jp = valid ? pt : 0;
  if (hf == 0) mask[m] = valid;
  for (int cc = hf; cc < K0; cc += 2) {
    float v = 0.f;
    if (valid) {
      if (cc < CX)       v = xb[(size_t)jp * CX + cc];
      else if (cc < CIN) v = pb[jp * 3 + (cc - CX)];
    }
    buf[(size_t)m * STRIDE + cc] = (h16)v;
  }
  __syncthreads();
  mlp_layer<K0, C1, STRIDE>(buf, rowBase, wt0, b0, g0, e0, lane);
  mlp_layer<C1, C2, STRIDE>(buf, rowBase, wt1, b1, g1, e1, lane);
  mlp_final_max<C2, C3, STRIDE>(buf, rowBase, wt2, b2, g2, e2, mask, red, lane, wave);
  __syncthreads();
  for (int ci = tid; ci < C3; ci += 64)
    partial[((size_t)b * nblk + blk) * C3 + ci] = fmaxf(red[ci], red[C3 + ci]);
}

__global__ void max_blocks_kernel(const float* __restrict__ part, float* __restrict__ out, int nblk, int C) {
  int i = blockIdx.x * blockDim.x + threadIdx.x;
  int b = i / C, c = i % C;
  float v = -INFINITY;
  for (int k = 0; k < nblk; ++k) v = fmaxf(v, part[((size_t)b * nblk + k) * C + c]);
  out[i] = v;
}

// ---------------- Weight transpose + f16 conversion (K x N f32 -> Npad x Kpad f16) ----------------
__global__ void conv_wt_kernel(const float* __restrict__ W, h16* __restrict__ WT,
                               int Kc, int Nc, int Kp, int Np) {
  int i = blockIdx.x * blockDim.x + threadIdx.x;
  if (i >= Kp * Np) return;
  int n = i / Kp, k = i % Kp;
  float v = (k < Kc && n < Nc) ? W[(size_t)k * Nc + n] : 0.f;
  WT[i] = (h16)v;
}

// ---------------- Front: split + fc_start/bn/leaky(0.2) + mlp1/leaky(0.2) ----------------
__global__ void point_front_kernel(const float* __restrict__ data, float* __restrict__ coords,
                                   float* __restrict__ x0, float* __restrict__ x1,
                                   const float* Wf, const float* bf, const float* gb, const float* bb,
                                   const float* Wm, const float* bm) {
  int i = blockIdx.x * blockDim.x + threadIdx.x;
  if (i >= B_ * N_) return;
  const float* dr = data + (size_t)i * 11;
  coords[i * 3 + 0] = dr[0]; coords[i * 3 + 1] = dr[1]; coords[i * 3 + 2] = dr[2];
  float lf[8], t[8];
#pragma unroll
  for (int k = 0; k < 8; ++k) lf[k] = dr[3 + k];
#pragma unroll
  for (int o = 0; o < 8; ++o) {
    float s = bf[o];
#pragma unroll
    for (int k = 0; k < 8; ++k) s += lf[k] * Wf[k * 8 + o];
    t[o] = leakyf(s * gb[o] + bb[o], 0.2f);
    x0[(size_t)i * 8 + o] = t[o];
  }
#pragma unroll
  for (int o = 0; o < 8; ++o) {
    float s = bm[o];
#pragma unroll
    for (int k = 0; k < 8; ++k) s += t[k] * Wm[k * 8 + o];
    x1[(size_t)i * 8 + o] = leakyf(s, 0.2f);
  }
}

// ---------------- KNN top-16 (per-wave, lexicographic (d2,idx) selection) ----------------
__global__ __launch_bounds__(32) void knn_kernel(const float* __restrict__ coords,
                                                 int* __restrict__ kidx, float* __restrict__ kdist) {
  __shared__ float ld[16 * 32];
  __shared__ int   li[16 * 32];
  int lane = threadIdx.x, pt = blockIdx.x;
  int base = pt & ~(N_ - 1), i = pt & (N_ - 1);
  const float* cb = coords + (size_t)base * 3;
  float cx = cb[i * 3], cy = cb[i * 3 + 1], cz = cb[i * 3 + 2];
  for (int p = 0; p < 16; ++p) { ld[p * 32 + lane] = INFINITY; li[p * 32 + lane] = 0x7fffffff; }
  for (int jj = lane; jj < N_; jj += 32) {
    float dx = cb[jj * 3] - cx, dy = cb[jj * 3 + 1] - cy, dz = cb[jj * 3 + 2] - cz;
    float d2 = dx * dx + dy * dy + dz * dz;
    float w = ld[15 * 32 + lane]; int wi = li[15 * 32 + lane];
    if (d2 < w || (d2 == w && jj < wi)) {
      int p = 15;
      while (p > 0) {
        float pv = ld[(p - 1) * 32 + lane]; int pi = li[(p - 1) * 32 + lane];
        if (d2 < pv || (d2 == pv && jj < pi)) { ld[p * 32 + lane] = pv; li[p * 32 + lane] = pi; --p; }
        else break;
      }
      ld[p * 32 + lane] = d2; li[p * 32 + lane] = jj;
    }
  }
  int head = 0;
  for (int r = 0; r < 16; ++r) {
    float v = head < 16 ? ld[head * 32 + lane] : INFINITY;
    int idx = head < 16 ? li[head * 32 + lane] : 0x7fffffff;
    int src = lane;
#pragma unroll
    for (int off = 16; off > 0; off >>= 1) {
      float ov = __shfl_xor(v, off, 32); int oi = __shfl_xor(idx, off, 32); int os = __shfl_xor(src, off, 32);
      if (ov < v || (ov == v && oi < idx)) { v = ov; idx = oi; src = os; }
    }
    if (lane == src) head++;
    if (lane == 0) { kidx[(size_t)pt * 16 + r] = idx; kdist[(size_t)pt * 16 + r] = sqrtf(fmaxf(v, 1e-12f)); }
  }
}

// ---------------- Fused LSE + attention pool (one wave per point) ----------------
__global__ __launch_bounds__(32) void lse_pool_kernel(
    const float* __restrict__ coords, const int* __restrict__ kidx, const float* __restrict__ kdist,
    const float* __restrict__ feats,
    const float* Wl, const float* bl, const float* gl, const float* el,
    const float* Ws,
    const float* Wp, const float* bp, const float* gp, const float* ep, int outC,
    float* __restrict__ out) {
  __shared__ float xt[16][16];
  __shared__ float fb[16];
  int lane = threadIdx.x, pt = blockIdx.x;
  int part = lane >> 4, n = lane & 15;
  float cx = coords[pt * 3], cy = coords[pt * 3 + 1], cz = coords[pt * 3 + 2];
  int ii = kidx[(size_t)pt * 16 + n];
  if (part == 0) {
    int base = pt & ~(N_ - 1);
    const float* nc = coords + (size_t)(base + ii) * 3;
    float nx = nc[0], ny = nc[1], nz = nc[2];
    float dd = kdist[(size_t)pt * 16 + n];
    float cat[10] = { cx, cy, cz, nx, ny, nz, cx - nx, cy - ny, cz - nz, dd };
#pragma unroll
    for (int jc = 0; jc < 8; ++jc) {
      float s = bl[jc];
#pragma unroll
      for (int k = 0; k < 10; ++k) s += cat[k] * Wl[k * 8 + jc];
      xt[n][jc] = fmaxf(s, 0.f) * gl[jc] + el[jc];
    }
  } else {
#pragma unroll
    for (int jc = 0; jc < 8; ++jc) xt[n][8 + jc] = feats[(size_t)pt * 8 + jc];
  }
  __syncthreads();
  if (lane < 16) {
    float y[16];
#pragma unroll
    for (int nn = 0; nn < 16; ++nn) {
      float s = 0.f;
#pragma unroll
      for (int k = 0; k < 16; ++k) s += xt[nn][k] * Ws[k * 16 + lane];
      y[nn] = s;
    }
    float mx = y[0];
#pragma unroll
    for (int nn = 1; nn < 16; ++nn) mx = fmaxf(mx, y[nn]);
    float ss = 0.f;
#pragma unroll
    for (int nn = 0; nn < 16; ++nn) { y[nn] = expf(y[nn] - mx); ss += y[nn]; }
    float f = 0.f;
#pragma unroll
    for (int nn = 0; nn < 16; ++nn) f += (y[nn] / ss) * xt[nn][lane];
    fb[lane] = f;
  }
  __syncthreads();
  if (lane < outC) {
    float s = bp[lane];
#pragma unroll
    for (int k = 0; k < 16; ++k) s += fb[k] * Wp[k * outC + lane];
    out[(size_t)pt * outC + lane] = fmaxf(s, 0.f) * gp[lane] + ep[lane];
  }
}

// ---------------- mlp2 + shortcut + leaky(0.01); then channel shuffle ----------------
__global__ void mlp2_kernel(const float* __restrict__ x0, const float* __restrict__ p2,
                            const float* Wsc, const float* bsc, const float* gsc, const float* esc,
                            const float* Wm, const float* bm, float* __restrict__ x2) {
  int i = blockIdx.x * blockDim.x + threadIdx.x;
  int pt = i >> 5, c = i & 31;
  float s = bsc[c];
#pragma unroll
  for (int k = 0; k < 8; ++k) s += x0[(size_t)pt * 8 + k] * Wsc[k * 32 + c];
  s = s * gsc[c] + esc[c];
  float v = bm[c] + s;
#pragma unroll
  for (int k = 0; k < 16; ++k) v += p2[(size_t)pt * 16 + k] * Wm[k * 32 + c];
  x2[i] = leakyf(v, 0.01f);
}

__global__ void shuffle_kernel(const float* __restrict__ x2, float* __restrict__ xs) {
  int i = blockIdx.x * blockDim.x + threadIdx.x;
  int b = i >> 17;               // N*32 = 2^17
  int rem = i & (131072 - 1);    // rem = n*32 + c
  int sn = rem & (N_ - 1), sc = rem >> 12;
  xs[i] = x2[((size_t)b * N_ + sn) * 32 + sc];
}

// ---------------- Farthest point sampling (one workgroup per batch) ----------------
__global__ __launch_bounds__(256) void fps_kernel(const float* __restrict__ pos, int nsrc, int S,
                                                  int* __restrict__ sel) {
  __shared__ float mind[4096];
  __shared__ float rv[256];
  __shared__ int   ri[256];
  int b = blockIdx.x, tid = threadIdx.x;
  const float* pb = pos + (size_t)b * nsrc * 3;
  float x0 = pb[0], y0 = pb[1], z0 = pb[2];
  for (int i = tid; i < nsrc; i += 256) {
    float dx = pb[i * 3] - x0, dy = pb[i * 3 + 1] - y0, dz = pb[i * 3 + 2] - z0;
    mind[i] = dx * dx + dy * dy + dz * dz;
  }
  if (tid == 0) sel[(size_t)b * S] = 0;
  __syncthreads();
  for (int it = 1; it < S; ++it) {
    float bv = -1.f; int bi = 0x7fffffff;
    for (int i = tid; i < nsrc; i += 256) {
      float v = mind[i];
      if (v > bv) { bv = v; bi = i; }
    }
    rv[tid] = bv; ri[tid] = bi;
    __syncthreads();
    for (int off = 128; off > 0; off >>= 1) {
      if (tid < off) {
        float ov = rv[tid + off]; int oi = ri[tid + off];
        if (ov > rv[tid] || (ov == rv[tid] && oi < ri[tid])) { rv[tid] = ov; ri[tid] = oi; }
      }
      __syncthreads();
    }
    int nxt = ri[0];
    if (tid == 0) sel[(size_t)b * S + it] = nxt;
    float px = pb[nxt * 3], py = pb[nxt * 3 + 1], pz = pb[nxt * 3 + 2];
    for (int i = tid; i < nsrc; i += 256) {
      float dx = pb[i * 3] - px, dy = pb[i * 3 + 1] - py, dz = pb[i * 3 + 2] - pz;
      mind[i] = fminf(mind[i], dx * dx + dy * dy + dz * dz);
    }
    __syncthreads();
  }
}

__global__ void gatherq_kernel(const float* __restrict__ pos, int nsrc, const int* __restrict__ sel,
                               int S, float* __restrict__ q, int total) {
  int i = blockIdx.x * blockDim.x + threadIdx.x;
  if (i >= total) return;
  int b = i / S;
  const float* p = pos + ((size_t)b * nsrc + sel[i]) * 3;
  q[i * 3 + 0] = p[0]; q[i * 3 + 1] = p[1]; q[i * 3 + 2] = p[2];
}

// ---------------- Ball query: 64 smallest d2 within radius; -1 = invalid ----------------
__global__ __launch_bounds__(32) void ballquery_kernel(const float* __restrict__ pos, int nsrc,
                                                       const float* __restrict__ q, int S, float r2,
                                                       int* __restrict__ nidx) {
  __shared__ float ld[64 * 32];
  __shared__ int   li[64 * 32];
  int lane = threadIdx.x, cen = blockIdx.x, b = cen / S;
  const float* pb = pos + (size_t)b * nsrc * 3;
  float qx = q[cen * 3], qy = q[cen * 3 + 1], qz = q[cen * 3 + 2];
  for (int p = 0; p < 64; ++p) { ld[p * 32 + lane] = INFINITY; li[p * 32 + lane] = 0x7fffffff; }
  for (int jj = lane; jj < nsrc; jj += 32) {
    float dx = pb[jj * 3] - qx, dy = pb[jj * 3 + 1] - qy, dz = pb[jj * 3 + 2] - qz;
    float d2 = dx * dx + dy * dy + dz * dz;
    if (d2 > r2) continue;
    float w = ld[63 * 32 + lane]; int wi = li[63 * 32 + lane];
    if (d2 < w || (d2 == w && jj < wi)) {
      int p = 63;
      while (p > 0) {
        float pv = ld[(p - 1) * 32 + lane]; int pi = li[(p - 1) * 32 + lane];
        if (d2 < pv || (d2 == pv && jj < pi)) { ld[p * 32 + lane] = pv; li[p * 32 + lane] = pi; --p; }
        else break;
      }
      ld[p * 32 + lane] = d2; li[p * 32 + lane] = jj;
    }
  }
  int head = 0;
  for (int r = 0; r < 64; ++r) {
    float v = head < 64 ? ld[head * 32 + lane] : INFINITY;
    int idx = head < 64 ? li[head * 32 + lane] : 0x7fffffff;
    int src = lane;
#pragma unroll
    for (int off = 16; off > 0; off >>= 1) {
      float ov = __shfl_xor(v, off, 32); int oi = __shfl_xor(idx, off, 32); int os = __shfl_xor(src, off, 32);
      if (ov < v || (ov == v && oi < idx)) { v = ov; idx = oi; src = os; }
    }
    if (lane == src) head++;
    if (lane == 0) nidx[(size_t)cen * 64 + r] = (v < INFINITY) ? idx : -1;
  }
}

// ---------------- Head GEMV + log-softmax ----------------
__global__ void linear_kernel(const float* __restrict__ in, const float* __restrict__ W,
                              const float* __restrict__ bias, float* __restrict__ out,
                              int IC, int OC, int relu) {
  int o = blockIdx.x * blockDim.x + threadIdx.x;
  int b = blockIdx.y;
  if (o >= OC) return;
  float s = bias[o];
  for (int k = 0; k < IC; ++k) s += in[(size_t)b * IC + k] * W[(size_t)k * OC + o];
  if (relu) s = fmaxf(s, 0.f);
  out[(size_t)b * OC + o] = s;
}

__global__ void logsoftmax2_kernel(const float* __restrict__ z, float* __restrict__ out) {
  int b = threadIdx.x;
  if (b < B_) {
    float z0 = z[b * 2], z1 = z[b * 2 + 1];
    float m = fmaxf(z0, z1);
    float l = m + logf(expf(z0 - m) + expf(z1 - m));
    out[b * 2] = z0 - l; out[b * 2 + 1] = z1 - l;
  }
}

// =====================================================================================
extern "C" void kernel_launch(void* const* d_in, const int* in_sizes, int n_in,
                              void* d_out, int out_size, void* d_ws, size_t ws_size,
                              hipStream_t stream) {
  (void)in_sizes; (void)n_in; (void)out_size; (void)ws_size;
  const float* data = (const float*)d_in[0];
  auto F = [&](int i) { return (const float*)d_in[i]; };

  char* ws = (char*)d_ws;
  size_t off = 0;
  auto alloc = [&](size_t bytes) -> void* {
    void* p = ws + off;
    off = (off + bytes + 255) & ~(size_t)255;
    return p;
  };
  auto allocF = [&](size_t n) { return (float*)alloc(n * 4); };
  auto allocI = [&](size_t n) { return (int*)alloc(n * 4); };
  auto allocH = [&](size_t n) { return (h16*)alloc(n * 2); };

  float* coords = allocF((size_t)B_ * N_ * 3);
  float* x0    = allocF((size_t)B_ * N_ * 8);
  float* x1    = allocF((size_t)B_ * N_ * 8);
  int*   kidx  = allocI((size_t)B_ * N_ * KNN_);
  float* kdist = allocF((size_t)B_ * N_ * KNN_);
  float* pool1 = allocF((size_t)B_ * N_ * 8);
  float* pool2 = allocF((size_t)B_ * N_ * 16);
  float* x2    = allocF((size_t)B_ * N_ * 32);
  float* xsh   = allocF((size_t)B_ * N_ * 32);
  int*   fps1  = allocI((size_t)B_ * S1_);
  float* q1    = allocF((size_t)B_ * S1_ * 3);
  int*   nidx1 = allocI((size_t)B_ * S1_ * NBR_);
  float* o1    = allocF((size_t)B_ * S1_ * 96);
  int*   fps2  = allocI((size_t)B_ * S1A_);
  float* q2    = allocF((size_t)B_ * S1A_ * 3);
  int*   nidx2 = allocI((size_t)B_ * S1A_ * NBR_);
  float* o2    = allocF((size_t)B_ * S1A_ * 128);
  int*   fps3  = allocI((size_t)B_ * S2_);
  float* q3    = allocF((size_t)B_ * S2_ * 3);
  int*   nidx3 = allocI((size_t)B_ * S2_ * NBR_);
  float* o3    = allocF((size_t)B_ * S2_ * 256);
  float* part3 = allocF((size_t)B_ * 4 * 1024);
  float* g3    = allocF((size_t)B_ * 1024);
  float* h1    = allocF((size_t)B_ * 512);
  float* h2    = allocF((size_t)B_ * 256);
  float* h3    = allocF((size_t)B_ * 128);
  float* zz    = allocF((size_t)B_ * 2);

  h16* w10 = allocH(64 * 64),    *w11 = allocH(64 * 64),    *w12 = allocH(96 * 64);
  h16* w20 = allocH(96 * 128),   *w21 = allocH(96 * 96),    *w22 = allocH(128 * 96);
  h16* w30 = allocH(128 * 160),  *w31 = allocH(128 * 128),  *w32 = allocH(256 * 128);
  h16* w40 = allocH(256 * 288),  *w41 = allocH(512 * 256),  *w42 = allocH(1024 * 512);

  auto cwt = [&](const float* W, h16* WT, int Kc, int Nc, int Kp, int Np) {
    int tot = Kp * Np;
    conv_wt_kernel<<<(tot + 255) / 256, 256, 0, stream>>>(W, WT, Kc, Nc, Kp, Np);
  };
  // sa1 [35,64,64,96]
  cwt(F(31), w10, 35, 64, 64, 64);  cwt(F(35), w11, 64, 64, 64, 64);  cwt(F(39), w12, 64, 96, 64, 96);
  // sa1a [99,96,96,128]
  cwt(F(43), w20, 99, 96, 128, 96); cwt(F(47), w21, 96, 96, 96, 96);  cwt(F(51), w22, 96, 128, 96, 128);
  // sa2 [131,128,128,256]
  cwt(F(55), w30, 131, 128, 160, 128); cwt(F(59), w31, 128, 128, 128, 128); cwt(F(63), w32, 128, 256, 128, 256);
  // sa3 [259,256,512,1024]
  cwt(F(67), w40, 259, 256, 288, 256); cwt(F(71), w41, 256, 512, 256, 512); cwt(F(75), w42, 512, 1024, 512, 1024);

  int npts = B_ * N_;
  point_front_kernel<<<npts / 256, 256, 0, stream>>>(data, coords, x0, x1,
      F(1), F(2), F(3), F(4), F(5), F(6));
  knn_kernel<<<npts, 32, 0, stream>>>(coords, kidx, kdist);
  lse_pool_kernel<<<npts, 32, 0, stream>>>(coords, kidx, kdist, x1,
      F(7), F(8), F(9), F(10), F(15), F(16), F(17), F(18), F(19), 8, pool1);
  lse_pool_kernel<<<npts, 32, 0, stream>>>(coords, kidx, kdist, pool1,
      F(11), F(12), F(13), F(14), F(20), F(21), F(22), F(23), F(24), 16, pool2);
  mlp2_kernel<<<npts * 32 / 256, 256, 0, stream>>>(x0, pool2,
      F(27), F(28), F(29), F(30), F(25), F(26), x2);
  shuffle_kernel<<<npts * 32 / 256, 256, 0, stream>>>(x2, xsh);

  // SA stage 1: coords(4096) -> 820 centers, r=0.2
  fps_kernel<<<B_, 256, 0, stream>>>(coords, N_, S1_, fps1);
  gatherq_kernel<<<(B_ * S1_ + 255) / 256, 256, 0, stream>>>(coords, N_, fps1, S1_, q1, B_ * S1_);
  ballquery_kernel<<<B_ * S1_, 32, 0, stream>>>(coords, N_, q1, S1_, 0.04f, nidx1);
  sa_kernel<32, 35, 64, 64, 64, 96, 64><<<B_ * S1_, 128, 0, stream>>>(
      xsh, coords, N_, q1, nidx1, S1_,
      w10, F(32), F(33), F(34), w11, F(36), F(37), F(38), w12, F(40), F(41), F(42), o1);

  // SA stage 1a: 820 -> 410, r=0.2
  fps_kernel<<<B_, 256, 0, stream>>>(q1, S1_, S1A_, fps2);
  gatherq_kernel<<<(B_ * S1A_ + 255) / 256, 256, 0, stream>>>(q1, S1_, fps2, S1A_, q2, B_ * S1A_);
  ballquery_kernel<<<B_ * S1A_, 32, 0, stream>>>(q1, S1_, q2, S1A_, 0.04f, nidx2);
  sa_kernel<96, 99, 128, 96, 96, 128, 128><<<B_ * S1A_, 128, 0, stream>>>(
      o1, q1, S1_, q2, nidx2, S1A_,
      w20, F(44), F(45), F(46), w21, F(48), F(49), F(50), w22, F(52), F(53), F(54), o2);

  // SA stage 2: 410 -> 103, r=0.4
  fps_kernel<<<B_, 256, 0, stream>>>(q2, S1A_, S2_, fps3);
  gatherq_kernel<<<(B_ * S2_ + 255) / 256, 256, 0, stream>>>(q2, S1A_, fps3, S2_, q3, B_ * S2_);
  ballquery_kernel<<<B_ * S2_, 32, 0, stream>>>(q2, S1A_, q3, S2_, 0.16f, nidx3);
  sa_kernel<128, 131, 160, 128, 128, 256, 160><<<B_ * S2_, 128, 0, stream>>>(
      o2, q2, S1A_, q3, nidx3, S2_,
      w30, F(56), F(57), F(58), w31, F(60), F(61), F(62), w32, F(64), F(65), F(66), o3);

  // Global SA: 103 points -> 1024 features (4 blocks of 32 rows per batch)
  gsa_kernel<256, 259, 288, 256, 512, 1024, 512><<<B_ * 4, 64, 0, stream>>>(
      o3, q3, S2_, S2_,
      w40, F(68), F(69), F(70), w41, F(72), F(73), F(74), w42, F(76), F(77), F(78), part3, 4);
  max_blocks_kernel<<<(B_ * 1024) / 256, 256, 0, stream>>>(part3, g3, 4, 1024);

  // Head
  linear_kernel<<<dim3(8, B_), 64, 0, stream>>>(g3, F(79), F(80), h1, 1024, 512, 1);
  linear_kernel<<<dim3(4, B_), 64, 0, stream>>>(h1, F(81), F(82), h2, 512, 256, 1);
  linear_kernel<<<dim3(2, B_), 64, 0, stream>>>(h2, F(83), F(84), h3, 256, 128, 1);
  linear_kernel<<<dim3(1, B_), 64, 0, stream>>>(h3, F(85), F(86), zz, 128, 2, 0);
  logsoftmax2_kernel<<<1, 32, 0, stream>>>(zz, (float*)d_out);
}
